// SplineLinear_65618510349022
// MI455X (gfx1250) — compile-verified, matrix operand order checked by reading, not run on hardware
//
#include <hip/hip_runtime.h>
#include <hip/hip_bf16.h>
#include <stdint.h>

typedef __attribute__((ext_vector_type(16))) __bf16 v16bf;
typedef __attribute__((ext_vector_type(8)))  __bf16 v8bf;
typedef __attribute__((ext_vector_type(4)))  __bf16 v4bf;
typedef __attribute__((ext_vector_type(8)))  float  v8f;
typedef __attribute__((ext_vector_type(4)))  int    v4i;

#if defined(__has_builtin)
#if __has_builtin(__builtin_amdgcn_global_load_async_to_lds_b128)
#define HAVE_ASYNC_LDS 1
#endif
#if __has_builtin(__builtin_amdgcn_s_wait_asynccnt)
#define HAVE_WAIT_ASYNC 1
#endif
#endif

namespace {
constexpr int kBatch = 16384;
constexpr int kInF   = 2048;
constexpr int kOutF  = 2048;
constexpr int kKspl  = kInF * 5;       // 10240
constexpr int kKtot  = kInF + kKspl;   // 12288 packed K (base | spline)

constexpr int BM = 128;
constexpr int BN = 128;                // main (packed-weight) kernel
constexpr int S1 = 40;                 // phase-1 LDS row stride (bf16), 80 B
constexpr int S2 = 168;                // phase-2 LDS row stride (bf16), 336 B
}

// Fast silu: result is truncated to bf16, so v_rcp_f32 (1 ULP) is exact enough;
// avoids the precise-division v_div_scale/v_fma Newton iteration chain.
__device__ __forceinline__ float siluf(float v) {
  return v * __builtin_amdgcn_rcpf(1.0f + __expf(-v));
}

// 16B global->LDS copy: async DMA path on CDNA5 (ASYNCcnt), plain copy fallback.
__device__ __forceinline__ void cp16_g2l(const __bf16* g, __bf16* l) {
#if defined(HAVE_ASYNC_LDS)
  __builtin_amdgcn_global_load_async_to_lds_b128(
      (__attribute__((address_space(1))) v4i*)(uintptr_t)g,  // flat==AS1 for global
      (__attribute__((address_space(3))) v4i*)l,             // addrspacecast to LDS
      0, 0);
#else
  *(uint4*)l = *(const uint4*)g;
#endif
}

__device__ __forceinline__ void async_fence() {
#if defined(HAVE_ASYNC_LDS)
#if defined(HAVE_WAIT_ASYNC)
  __builtin_amdgcn_s_wait_asynccnt(0);
#else
  asm volatile("s_wait_asynccnt 0" ::: "memory");
#endif
#endif
}

// 16x32 bf16 WMMA operand fragment from a row-major LDS tile
// (CDNA5 ISA: lane l -> row l%16; elems 0..7 = K kh..kh+7,
//  elems 8..15 = K 16+kh..16+kh+7, kh=(l/16)*8 -> two ds_load_b128 runs).
__device__ __forceinline__ v16bf frag_ld(const __bf16* base, int stride, int lane) {
  const int m  = lane & 15;
  const int kh = (lane >> 4) << 3;
  const __bf16* p = base + m * stride;
  v8bf lo = *(const v8bf*)(p + kh);
  v8bf hi = *(const v8bf*)(p + 16 + kh);
  return __builtin_shufflevector(lo, hi, 0,1,2,3,4,5,6,7,8,9,10,11,12,13,14,15);
}

// ---- tile producers -------------------------------------------------------
__device__ __forceinline__ void issueB1(const __bf16* wp, int o0, int kt,
                                        __bf16* sBb, int tid) {
  #pragma unroll
  for (int p = 0; p < 2; ++p) {
    const int f = tid + p * 256, row = f >> 2, ce = (f & 3) * 8;
    cp16_g2l(wp + (size_t)(o0 + row) * kKtot + kt + ce, sBb + row * S1 + ce);
  }
}

__device__ __forceinline__ void produceA1(const float* x, int b0, int kt,
                                          __bf16* sAb, int tid) {
  #pragma unroll
  for (int p = 0; p < 4; ++p) {
    const int f = tid + p * 256, row = f >> 3, c4 = (f & 7) << 2;
    const float4 xv = *(const float4*)(x + (size_t)(b0 + row) * kInF + kt + c4);
    v4bf a;
    a[0] = (__bf16)siluf(xv.x); a[1] = (__bf16)siluf(xv.y);
    a[2] = (__bf16)siluf(xv.z); a[3] = (__bf16)siluf(xv.w);
    *(v4bf*)(sAb + row * S1 + c4) = a;
  }
}

__device__ __forceinline__ void issueB2(const __bf16* wp, int o0, int t0,
                                        __bf16* sBb, int tid) {
  #pragma unroll
  for (int p = 0; p < 10; ++p) {
    const int f = tid + p * 256, row = f / 20, ce = (f - row * 20) * 8;
    cp16_g2l(wp + (size_t)(o0 + row) * kKtot + kInF + t0 + ce, sBb + row * S2 + ce);
  }
}

__device__ __forceinline__ void produceA2(const float* x, int b0, int i0,
                                          __bf16* sAb, int tid) {
  #pragma unroll
  for (int p = 0; p < 4; ++p) {
    const int f = tid + p * 256, row = f >> 3, i4 = (f & 7) << 2;
    const float4 xv = *(const float4*)(x + (size_t)(b0 + row) * kInF + i0 + i4);
    const float xs[4] = {xv.x, xv.y, xv.z, xv.w};
    __bf16* dst = sAb + row * S2 + i4 * 5;
    #pragma unroll
    for (int q = 0; q < 5; ++q) {
      v4bf t;
      #pragma unroll
      for (int e = 0; e < 4; ++e) {
        const int idx = q * 4 + e, j = idx / 5, g = idx - 5 * j;
        const float d = xs[j] - (-1.0f + 0.5f * (float)g);
        t[e] = (__bf16)__expf(-5.0f * d * d);
      }
      *(v4bf*)(dst + 4 * q) = t;
    }
  }
}

// ---------------------------------------------------------------------------
// Pack [base_weight | spline_weights] (f32) -> bf16 [2048 x 12288], row o.
// ---------------------------------------------------------------------------
__global__ __launch_bounds__(256)
void pack_weights(const float* __restrict__ bw, const float* __restrict__ sw,
                  __bf16* __restrict__ wp) {
  const size_t f = ((size_t)blockIdx.x * 256 + threadIdx.x) * 8;  // elem index
  const int o = (int)(f / kKtot);
  const int k = (int)(f % kKtot);
  const float* src = (k < kInF) ? (bw + (size_t)o * kInF + k)
                                : (sw + (size_t)o * kKspl + (k - kInF));
  const float4 a = *(const float4*)src;
  const float4 b = *(const float4*)(src + 4);
  v8bf v;
  v[0] = (__bf16)a.x; v[1] = (__bf16)a.y; v[2] = (__bf16)a.z; v[3] = (__bf16)a.w;
  v[4] = (__bf16)b.x; v[5] = (__bf16)b.y; v[6] = (__bf16)b.z; v[7] = (__bf16)b.w;
  *(v8bf*)(wp + f) = v;
}

// ---------------------------------------------------------------------------
// Main fused kernel: 128x128 tile, 8 wave32 in 4x2, 32x64 (2x4 tiles) per wave,
// double-buffered LDS with async weight stream pipelined against WMMA.
// ---------------------------------------------------------------------------
__global__ __launch_bounds__(256)
void spline_linear_wmma(const float* __restrict__ x,
                        const __bf16* __restrict__ wp,
                        float* __restrict__ out) {
  __shared__ __align__(16) __bf16 sA[2][BM * S2];  // 2 x 43008 B
  __shared__ __align__(16) __bf16 sB[2][BN * S2];  // 2 x 43008 B

  const int tid  = threadIdx.x;
  const int lane = tid & 31;
  const int wid  = tid >> 5;
  const int wm   = (wid & 3) * 32;
  const int wn   = (wid >> 2) * 64;

  const int b0 = blockIdx.y * BM;
  const int o0 = blockIdx.x * BN;

  v8f acc[2][4];
  #pragma unroll
  for (int mi = 0; mi < 2; ++mi)
    #pragma unroll
    for (int ni = 0; ni < 4; ++ni)
      acc[mi][ni] = (v8f){0.f, 0.f, 0.f, 0.f, 0.f, 0.f, 0.f, 0.f};

  // -------- Phase 1: base path, K = 2048, BK = 32, pipelined ---------------
  issueB1(wp, o0, 0, sB[0], tid);
  produceA1(x, b0, 0, sA[0], tid);
  async_fence();
  __syncthreads();
  for (int it = 0; it < kInF / 32; ++it) {
    const int cur = it & 1, nxt = cur ^ 1;
    const int ktn = (it + 1) * 32;
    if (it + 1 < kInF / 32) issueB1(wp, o0, ktn, sB[nxt], tid);  // async, no wait

    v16bf af[2], bfr[4];
    #pragma unroll
    for (int mi = 0; mi < 2; ++mi) af[mi]  = frag_ld(sA[cur] + (wm + mi * 16) * S1, S1, lane);
    #pragma unroll
    for (int ni = 0; ni < 4; ++ni) bfr[ni] = frag_ld(sB[cur] + (wn + ni * 16) * S1, S1, lane);
    #pragma unroll
    for (int mi = 0; mi < 2; ++mi)
      #pragma unroll
      for (int ni = 0; ni < 4; ++ni)
        acc[mi][ni] = __builtin_amdgcn_wmma_f32_16x16x32_bf16(
            false, af[mi], false, bfr[ni], (short)0, acc[mi][ni], false, false);

    if (it + 1 < kInF / 32) produceA1(x, b0, ktn, sA[nxt], tid); // VALU overlaps WMMA
    async_fence();
    __syncthreads();
  }

  // -------- Phase 2: spline path, K-tile = 160 = 32 i x 5 g, pipelined -----
  issueB2(wp, o0, 0, sB[0], tid);
  produceA2(x, b0, 0, sA[0], tid);
  async_fence();
  __syncthreads();
  for (int jt = 0; jt < kInF / 32; ++jt) {
    const int cur = jt & 1, nxt = cur ^ 1;
    if (jt + 1 < kInF / 32) issueB2(wp, o0, (jt + 1) * 160, sB[nxt], tid);

    #pragma unroll
    for (int kk = 0; kk < 5; ++kk) {
      const int kb = kk * 32;
      v16bf af[2], bfr[4];
      #pragma unroll
      for (int mi = 0; mi < 2; ++mi) af[mi]  = frag_ld(sA[cur] + (wm + mi * 16) * S2 + kb, S2, lane);
      #pragma unroll
      for (int ni = 0; ni < 4; ++ni) bfr[ni] = frag_ld(sB[cur] + (wn + ni * 16) * S2 + kb, S2, lane);
      #pragma unroll
      for (int mi = 0; mi < 2; ++mi)
        #pragma unroll
        for (int ni = 0; ni < 4; ++ni)
          acc[mi][ni] = __builtin_amdgcn_wmma_f32_16x16x32_bf16(
              false, af[mi], false, bfr[ni], (short)0, acc[mi][ni], false, false);
    }

    if (jt + 1 < kInF / 32) produceA2(x, b0, (jt + 1) * 32, sA[nxt], tid);
    async_fence();
    __syncthreads();
  }

  // -------- Epilogue -------------------------------------------------------
  const int lm = lane & 15;
  const int lh = lane >> 4;
  #pragma unroll
  for (int mi = 0; mi < 2; ++mi)
    #pragma unroll
    for (int ni = 0; ni < 4; ++ni)
      #pragma unroll
      for (int v = 0; v < 8; ++v) {
        const int r = b0 + wm + mi * 16 + v + 8 * lh;
        const int c = o0 + wn + ni * 16 + lm;
        out[(size_t)r * kOutF + c] = acc[mi][ni][v];
      }
}

// ---------------------------------------------------------------------------
// Fallback (no-workspace) kernel: f32 weights converted inline, single-buffered.
// ---------------------------------------------------------------------------
__global__ __launch_bounds__(256)
void spline_linear_wmma_fb(const float* __restrict__ x,
                           const float* __restrict__ bw,
                           const float* __restrict__ sw,
                           float* __restrict__ out) {
  __shared__ __align__(16) __bf16 sA[128 * S2];
  __shared__ __align__(16) __bf16 sB[64 * S2];

  const int tid  = threadIdx.x;
  const int lane = tid & 31;
  const int wid  = tid >> 5;
  const int wm   = (wid & 3) * 32;
  const int wn   = (wid >> 2) * 32;
  const int b0 = blockIdx.y * 128;
  const int o0 = blockIdx.x * 64;

  v8f acc[2][2];
  #pragma unroll
  for (int mi = 0; mi < 2; ++mi)
    #pragma unroll
    for (int ni = 0; ni < 2; ++ni)
      acc[mi][ni] = (v8f){0.f, 0.f, 0.f, 0.f, 0.f, 0.f, 0.f, 0.f};

  for (int kt = 0; kt < kInF; kt += 32) {
    produceA1(x, b0, kt, sA, tid);
    #pragma unroll
    for (int p = 0; p < 2; ++p) {
      const int f = tid + p * 256, row = f >> 3, c4 = (f & 7) << 2;
      const float4 wv = *(const float4*)(bw + (size_t)(o0 + row) * kInF + kt + c4);
      v4bf b;
      b[0] = (__bf16)wv.x; b[1] = (__bf16)wv.y; b[2] = (__bf16)wv.z; b[3] = (__bf16)wv.w;
      *(v4bf*)(sB + row * S1 + c4) = b;
    }
    __syncthreads();
    v16bf af[2], bfr[2];
    #pragma unroll
    for (int mi = 0; mi < 2; ++mi) af[mi]  = frag_ld(sA + (wm + mi * 16) * S1, S1, lane);
    #pragma unroll
    for (int ni = 0; ni < 2; ++ni) bfr[ni] = frag_ld(sB + (wn + ni * 16) * S1, S1, lane);
    #pragma unroll
    for (int mi = 0; mi < 2; ++mi)
      #pragma unroll
      for (int ni = 0; ni < 2; ++ni)
        acc[mi][ni] = __builtin_amdgcn_wmma_f32_16x16x32_bf16(
            false, af[mi], false, bfr[ni], (short)0, acc[mi][ni], false, false);
    __syncthreads();
  }

  for (int jt = 0; jt < kInF / 32; ++jt) {
    const int i0 = jt * 32, t0 = jt * 160;
    produceA2(x, b0, i0, sA, tid);
    #pragma unroll
    for (int p = 0; p < 10; ++p) {
      const int f = tid + p * 256;
      const int row = f / 40;
      const int c4 = (f - row * 40) * 4;
      const float4 wv = *(const float4*)(sw + (size_t)(o0 + row) * kKspl + t0 + c4);
      v4bf b;
      b[0] = (__bf16)wv.x; b[1] = (__bf16)wv.y; b[2] = (__bf16)wv.z; b[3] = (__bf16)wv.w;
      *(v4bf*)(sB + row * S2 + c4) = b;
    }
    __syncthreads();
    #pragma unroll
    for (int kk = 0; kk < 5; ++kk) {
      const int kb = kk * 32;
      v16bf af[2], bfr[2];
      #pragma unroll
      for (int mi = 0; mi < 2; ++mi) af[mi]  = frag_ld(sA + (wm + mi * 16) * S2 + kb, S2, lane);
      #pragma unroll
      for (int ni = 0; ni < 2; ++ni) bfr[ni] = frag_ld(sB + (wn + ni * 16) * S2 + kb, S2, lane);
      #pragma unroll
      for (int mi = 0; mi < 2; ++mi)
        #pragma unroll
        for (int ni = 0; ni < 2; ++ni)
          acc[mi][ni] = __builtin_amdgcn_wmma_f32_16x16x32_bf16(
              false, af[mi], false, bfr[ni], (short)0, acc[mi][ni], false, false);
    }
    __syncthreads();
  }

  const int lm = lane & 15, lh = lane >> 4;
  #pragma unroll
  for (int mi = 0; mi < 2; ++mi)
    #pragma unroll
    for (int ni = 0; ni < 2; ++ni)
      #pragma unroll
      for (int v = 0; v < 8; ++v) {
        const int r = b0 + wm + mi * 16 + v + 8 * lh;
        const int c = o0 + wn + ni * 16 + lm;
        out[(size_t)r * kOutF + c] = acc[mi][ni][v];
      }
}

extern "C" void kernel_launch(void* const* d_in, const int* in_sizes, int n_in,
                              void* d_out, int out_size, void* d_ws, size_t ws_size,
                              hipStream_t stream) {
  (void)in_sizes; (void)n_in; (void)out_size;
  const float* x  = (const float*)d_in[0];   // [16384, 2048] f32
  const float* bw = (const float*)d_in[1];   // [2048, 2048] f32
  const float* sw = (const float*)d_in[2];   // [2048, 2048, 5] f32
  float* out = (float*)d_out;                // [16384, 2048] f32

  const size_t packBytes = (size_t)kOutF * kKtot * 2;  // 50,331,648 B
  if (ws_size >= packBytes) {
    __bf16* wp = (__bf16*)d_ws;
    const int packBlocks = (kOutF * kKtot) / (256 * 8);  // 12288
    pack_weights<<<dim3(packBlocks), dim3(256), 0, stream>>>(bw, sw, wp);
    spline_linear_wmma<<<dim3(kOutF / BN, kBatch / BM), dim3(256), 0, stream>>>(x, wp, out);
  } else {
    spline_linear_wmma_fb<<<dim3(kOutF / 64, kBatch / BM), dim3(256), 0, stream>>>(x, bw, sw, out);
  }
}
